// WanAI2VCrossAttention_63711544869366
// MI455X (gfx1250) — matrix operand order repaired and checked by reading, not compiled, not run on hardware
//
#include <hip/hip_runtime.h>
#include <hip/hip_bf16.h>

// ---------------------------------------------------------------------------
// CDNA5 (gfx1250) wave32 WMMA implementation of WanAI2V cross-attention.
// Matrix math: V_WMMA_F32_16X16X32_BF16. GEMM staging: async global->LDS
// (double buffered, ASYNCcnt) when available, else load+ds_store fallback.
// ---------------------------------------------------------------------------

typedef __attribute__((ext_vector_type(16))) __bf16 v16bf;
typedef __attribute__((ext_vector_type(8)))  __bf16 v8bf;
typedef __attribute__((ext_vector_type(8)))  float  v8f;
typedef int v4i __attribute__((vector_size(16)));        // matches builtin param
typedef __attribute__((address_space(1))) v4i gv4i;      // global  (__device__)
typedef __attribute__((address_space(3))) v4i lv4i;      // LDS     (__shared__)

#define DIM   2048
#define NH    16
#define HD    128
#define L1    8192
#define LIMG  257
#define LTXT  512
#define LCTX  769

#if __has_builtin(__builtin_amdgcn_global_load_async_to_lds_b128)
#define HAVE_ASYNC_LDS 1
#else
#define HAVE_ASYNC_LDS 0
#endif

#if __has_builtin(__builtin_amdgcn_s_wait_asynccnt)
#define WAIT_ASYNC0() __builtin_amdgcn_s_wait_asynccnt(0)
#else
#define WAIT_ASYNC0() asm volatile("s_wait_asynccnt 0x0" ::: "memory")
#endif

static __device__ inline v16bf mk16(v8bf lo, v8bf hi) {
  v16bf r;
#pragma unroll
  for (int i = 0; i < 8; ++i) { r[i] = lo[i]; r[i + 8] = hi[i]; }
  return r;
}

static __device__ inline v8f zero8() {
  v8f r;
#pragma unroll
  for (int i = 0; i < 8; ++i) r[i] = 0.f;
  return r;
}

static __device__ inline v8bf zero8bf() {
  v8bf r;
#pragma unroll
  for (int i = 0; i < 8; ++i) r[i] = (__bf16)0.f;
  return r;
}

// 16-byte global -> LDS copy (async if available)
static __device__ inline void copy16_g2l(const __bf16* g, __bf16* l) {
#if HAVE_ASYNC_LDS
  __builtin_amdgcn_global_load_async_to_lds_b128(
      (gv4i*)(void*)g, (lv4i*)(void*)l, 0, 0);
#else
  *(v8bf*)l = *(const v8bf*)g;
#endif
}

static __device__ inline void wait_g2l() {
#if HAVE_ASYNC_LDS
  WAIT_ASYNC0();
#endif
}

// ---------------------------------------------------------------------------
// f32 -> bf16 flat convert
// ---------------------------------------------------------------------------
__global__ void cvt_f32_bf16_kernel(const float* __restrict__ in,
                                    __bf16* __restrict__ out, long long n) {
  long long i = (long long)blockIdx.x * blockDim.x + threadIdx.x;
  if (i < n) out[i] = (__bf16)in[i];
}

// ---------------------------------------------------------------------------
// W[K][N] f32  ->  WT[N][K] bf16   (32x32 LDS tile transpose)
// ---------------------------------------------------------------------------
__global__ void transpose_f32_bf16_kernel(const float* __restrict__ in,
                                          __bf16* __restrict__ out,
                                          int R, int C) {
  __shared__ __bf16 tile[32][33];
  int x = blockIdx.x * 32 + threadIdx.x;
#pragma unroll
  for (int i = 0; i < 4; ++i) {
    int y = blockIdx.y * 32 + threadIdx.y + i * 8;
    if (y < R && x < C) tile[threadIdx.y + i * 8][threadIdx.x] = (__bf16)in[(size_t)y * C + x];
  }
  __syncthreads();
  int r2 = blockIdx.y * 32 + threadIdx.x;   // input row  -> output col
#pragma unroll
  for (int i = 0; i < 4; ++i) {
    int c2 = blockIdx.x * 32 + threadIdx.y + i * 8;  // input col -> output row
    if (r2 < R && c2 < C) out[(size_t)c2 * R + r2] = tile[threadIdx.x][threadIdx.y + i * 8];
  }
}

// bf16 [R][C] -> bf16 [C][R]
__global__ void transpose_bf16_kernel(const __bf16* __restrict__ in,
                                      __bf16* __restrict__ out,
                                      int R, int C) {
  __shared__ __bf16 tile[32][33];
  int x = blockIdx.x * 32 + threadIdx.x;
#pragma unroll
  for (int i = 0; i < 4; ++i) {
    int y = blockIdx.y * 32 + threadIdx.y + i * 8;
    if (y < R && x < C) tile[threadIdx.y + i * 8][threadIdx.x] = in[(size_t)y * C + x];
  }
  __syncthreads();
  int r2 = blockIdx.y * 32 + threadIdx.x;
#pragma unroll
  for (int i = 0; i < 4; ++i) {
    int c2 = blockIdx.x * 32 + threadIdx.y + i * 8;
    if (r2 < R && c2 < C) out[(size_t)c2 * R + r2] = tile[threadIdx.x][threadIdx.y + i * 8];
  }
}

// ---------------------------------------------------------------------------
// C[M][N] f32 = A[M][K] bf16 @ WT[N][K] bf16 (+ bias[N])
// 256 threads = 8 waves, 128x128 block tile, K-step 32.
// Double-buffered LDS, async global->LDS staging, L2 prefetch 2 steps ahead.
// Wave (mg,ng): each wave owns a 32x64 sub-tile = 2x4 16x16 accumulators.
// ---------------------------------------------------------------------------
__global__ __launch_bounds__(256) void gemm_bias_kernel(
    const __bf16* __restrict__ A, const __bf16* __restrict__ WT,
    const float* __restrict__ bias, float* __restrict__ C,
    int M, int N, int K) {
  __shared__ __align__(16) __bf16 As[2][128 * 32];
  __shared__ __align__(16) __bf16 Bs[2][128 * 32];

  const int t    = threadIdx.x;
  const int lane = t & 31;
  const int wave = t >> 5;
  const int half = lane >> 4;
  const int lm   = lane & 15;
  const int mg   = wave >> 1;
  const int ng   = wave & 1;
  const int bm   = blockIdx.x * 128;
  const int bn   = blockIdx.y * 128;
  const int lr   = t >> 2;          // staging row 0..63
  const int lc   = (t & 3) * 8;     // staging col group (8 bf16 = 16B)
  const int nk   = K >> 5;

  v8f acc[2][4];
#pragma unroll
  for (int mi = 0; mi < 2; ++mi)
#pragma unroll
    for (int ni = 0; ni < 4; ++ni) acc[mi][ni] = zero8();

  auto stage = [&](int step, int buf) {
    const int k0 = step << 5;
#pragma unroll
    for (int rr = 0; rr < 2; ++rr) {
      int r    = lr + rr * 64;
      int grow = bm + r;
      __bf16* la = As[buf] + r * 32 + lc;
      if (grow < M) copy16_g2l(A + (size_t)grow * K + k0 + lc, la);
      else          *(v8bf*)la = zero8bf();
      int nrow = bn + r;  // N is always a multiple of 128 here
      copy16_g2l(WT + (size_t)nrow * K + k0 + lc, Bs[buf] + r * 32 + lc);
    }
  };

  stage(0, 0);
  wait_g2l();
  __syncthreads();

  for (int k = 0; k < nk; ++k) {
    const int buf = k & 1;
    if (k + 1 < nk) stage(k + 1, buf ^ 1);
    if (k + 2 < nk) {  // warm L2 two steps ahead
      const int kp = (k + 2) << 5;
      int grow = bm + lr;
      if (grow < M) __builtin_prefetch(A + (size_t)grow * K + kp + lc, 0, 0);
      __builtin_prefetch(WT + (size_t)(bn + lr) * K + kp + lc, 0, 0);
    }

    v16bf af[2], bf[4];
#pragma unroll
    for (int mi = 0; mi < 2; ++mi) {
      const __bf16* p = As[buf] + (mg * 32 + mi * 16 + lm) * 32 + half * 8;
      af[mi] = mk16(*(const v8bf*)p, *(const v8bf*)(p + 16));
    }
#pragma unroll
    for (int ni = 0; ni < 4; ++ni) {
      const __bf16* p = Bs[buf] + (ng * 64 + ni * 16 + lm) * 32 + half * 8;
      bf[ni] = mk16(*(const v8bf*)p, *(const v8bf*)(p + 16));
    }
#pragma unroll
    for (int mi = 0; mi < 2; ++mi)
#pragma unroll
      for (int ni = 0; ni < 4; ++ni)
        acc[mi][ni] = __builtin_amdgcn_wmma_f32_16x16x32_bf16(
            false, af[mi], false, bf[ni], (short)0, acc[mi][ni], false, false);

    wait_g2l();       // next tile landed (this wave)
    __syncthreads();  // all waves: tile k+1 visible, buf k free for k+2 staging
  }

#pragma unroll
  for (int mi = 0; mi < 2; ++mi)
#pragma unroll
    for (int ni = 0; ni < 4; ++ni)
#pragma unroll
      for (int v = 0; v < 8; ++v) {
        int row = bm + mg * 32 + mi * 16 + v + half * 8;
        int col = bn + ng * 64 + ni * 16 + lm;
        if (row < M) C[(size_t)row * N + col] = acc[mi][ni][v] + bias[col];
      }
}

// ---------------------------------------------------------------------------
// Row-wise RMSNorm (over DIM=2048) + convert to bf16; rows >= validRows get 0.
// One block (256 thr) per row, 8 elems/thread.
// ---------------------------------------------------------------------------
__global__ __launch_bounds__(256) void rms_cvt_kernel(
    const float* __restrict__ in, __bf16* __restrict__ out,
    const float* __restrict__ g, int validRows, int doRms) {
  __shared__ float red[8];
  int row = blockIdx.x;
  int t   = threadIdx.x;
  if (row >= validRows) {
#pragma unroll
    for (int i = 0; i < 8; ++i) out[(size_t)row * DIM + t + i * 256] = (__bf16)0.f;
    return;
  }
  float vals[8];
  float ss = 0.f;
#pragma unroll
  for (int i = 0; i < 8; ++i) {
    float v = in[(size_t)row * DIM + t + i * 256];
    vals[i] = v;
    ss += v * v;
  }
  float scale = 1.f;
  if (doRms) {
#pragma unroll
    for (int d = 1; d < 32; d <<= 1) ss += __shfl_xor(ss, d, 32);
    if ((t & 31) == 0) red[t >> 5] = ss;
    __syncthreads();
    float tot = 0.f;
#pragma unroll
    for (int i = 0; i < 8; ++i) tot += red[i];
    scale = rsqrtf(tot * (1.f / DIM) + 1e-6f);
  }
#pragma unroll
  for (int i = 0; i < 8; ++i) {
    int c    = t + i * 256;
    float gv = doRms ? g[c] : 1.f;
    out[(size_t)row * DIM + c] = (__bf16)(vals[i] * scale * gv);
  }
}

// ---------------------------------------------------------------------------
// RoPE on normalized q: 3D (f,h,w) angles from freqs[1024][64].
// one thread per (m, head, pair j).  s0 = 22, s1 = 21 (HD/2 = 64).
// ---------------------------------------------------------------------------
__global__ void rope_kernel(const __bf16* __restrict__ qn,
                            const float* __restrict__ freqs,
                            __bf16* __restrict__ qr) {
  int idx = blockIdx.x * blockDim.x + threadIdx.x;  // < 8192*16*64
  int j = idx & 63;
  int h = (idx >> 6) & 15;
  int m = idx >> 10;
  int f  = m >> 10;
  int hh = (m >> 5) & 31;
  int ww = m & 31;
  int p = (j < 22) ? f : ((j < 43) ? hh : ww);
  float a = freqs[p * 64 + j];
  float s, c;
  __sincosf(a, &s, &c);
  size_t base = (size_t)m * DIM + h * HD + 2 * j;
  float x0 = (float)qn[base], x1 = (float)qn[base + 1];
  qr[base]     = (__bf16)(x0 * c - x1 * s);
  qr[base + 1] = (__bf16)(x0 * s + x1 * c);
}

// ---------------------------------------------------------------------------
// Flash cross-attention, one wave per (16-query block, head).
// Q [8192][2048] bf16, K [LkPadRows][2048] bf16, VT [2048][VTcols] bf16.
// S = Q Kᵀ via 4x WMMA (K-dim 128), online softmax (16-lane shuffles),
// P staged through wave-private LDS to re-fragment, P·V via WMMA.
// mode: 0 = store, 1 = add, 2 = add * face_mask[row].
// audio != 0: key block = 32 keys of frame (qbase>>10).
// ---------------------------------------------------------------------------
__global__ __launch_bounds__(128) void flash_attn_kernel(
    const __bf16* __restrict__ Q, const __bf16* __restrict__ K,
    const __bf16* __restrict__ VT, float* __restrict__ Out,
    const float* __restrict__ face, const int* __restrict__ lens,
    int LkFixed, int LkIter, int VTcols, int mode, int audio) {
  __shared__ __align__(16) __bf16 Pl[4][16][32];

  const int tid  = threadIdx.x;
  const int lane = tid & 31;
  const int w    = tid >> 5;
  const int half = lane >> 4;
  const int lm   = lane & 15;
  const int qblock = blockIdx.x * 4 + w;
  const int qbase  = qblock * 16;
  const int hoff   = blockIdx.y * HD;

  int Lk = LkFixed;
  if (lens) { int L = lens[0]; if (L < Lk) Lk = L; }
  const int keyBase = audio ? ((qbase >> 10) << 5) : 0;

  const int kbA = half * 8;
  const int kbB = half * 8;

  v16bf qf[4];
#pragma unroll
  for (int s = 0; s < 4; ++s) {
    const __bf16* p = Q + (size_t)(qbase + lm) * DIM + hoff + 32 * s + kbA;
    qf[s] = mk16(*(const v8bf*)p, *(const v8bf*)(p + 16));
  }

  v8f acc[8];
#pragma unroll
  for (int ni = 0; ni < 8; ++ni) acc[ni] = zero8();
  float rmax[8], rsum[8];
#pragma unroll
  for (int v = 0; v < 8; ++v) { rmax[v] = -3.0e38f; rsum[v] = 0.f; }

  const float sc = 0.08838834764831845f;  // 1/sqrt(128)

  for (int kb = 0; kb < LkIter; kb += 32) {
    if (kb + 32 < LkIter) {  // warm L2 for the next key block
      __builtin_prefetch(K + (size_t)(keyBase + kb + 32 + lm) * DIM + hoff, 0, 0);
      __builtin_prefetch(VT + (size_t)(hoff + lm * 8) * VTcols + keyBase + kb + 32, 0, 0);
    }
    v8f s0 = zero8(), s1 = zero8();
#pragma unroll
    for (int s = 0; s < 4; ++s) {
      const __bf16* kp0 = K + (size_t)(keyBase + kb + lm) * DIM + hoff + 32 * s + kbB;
      v16bf b0 = mk16(*(const v8bf*)kp0, *(const v8bf*)(kp0 + 16));
      s0 = __builtin_amdgcn_wmma_f32_16x16x32_bf16(false, qf[s], false, b0, (short)0, s0, false, false);
      const __bf16* kp1 = K + (size_t)(keyBase + kb + 16 + lm) * DIM + hoff + 32 * s + kbB;
      v16bf b1 = mk16(*(const v8bf*)kp1, *(const v8bf*)(kp1 + 16));
      s1 = __builtin_amdgcn_wmma_f32_16x16x32_bf16(false, qf[s], false, b1, (short)0, s1, false, false);
    }
    const bool ok0 = (kb + lm) < Lk;
    const bool ok1 = (kb + 16 + lm) < Lk;
#pragma unroll
    for (int v = 0; v < 8; ++v) {
      float a0 = ok0 ? s0[v] * sc : -1.0e30f;
      float a1 = ok1 ? s1[v] * sc : -1.0e30f;
      float mx = fmaxf(a0, a1);
#pragma unroll
      for (int d = 1; d < 16; d <<= 1) mx = fmaxf(mx, __shfl_xor(mx, d, 32));
      float nm = fmaxf(rmax[v], mx);
      float so = __expf(rmax[v] - nm);
      float p0 = __expf(a0 - nm);
      float p1 = __expf(a1 - nm);
      float ps = p0 + p1;
#pragma unroll
      for (int d = 1; d < 16; d <<= 1) ps += __shfl_xor(ps, d, 32);
      rsum[v] = rsum[v] * so + ps;
      rmax[v] = nm;
#pragma unroll
      for (int ni = 0; ni < 8; ++ni) acc[ni][v] *= so;
      // Stage P tile (C-layout) into wave-private LDS for A-fragment reload.
      Pl[w][v + half * 8][lm]      = (__bf16)p0;
      Pl[w][v + half * 8][lm + 16] = (__bf16)p1;
    }
    const __bf16* pp = &Pl[w][lm][kbA];
    v16bf pf = mk16(*(const v8bf*)pp, *(const v8bf*)(pp + 16));
#pragma unroll
    for (int ni = 0; ni < 8; ++ni) {
      const __bf16* vp = VT + (size_t)(hoff + ni * 16 + lm) * VTcols + keyBase + kb + kbB;
      v16bf vf = mk16(*(const v8bf*)vp, *(const v8bf*)(vp + 16));
      acc[ni] = __builtin_amdgcn_wmma_f32_16x16x32_bf16(false, pf, false, vf, (short)0, acc[ni], false, false);
    }
  }

#pragma unroll
  for (int v = 0; v < 8; ++v) {
    float inv = 1.f / rsum[v];
    int row = qbase + v + half * 8;
    float fm = (mode == 2) ? face[row] : 1.f;
#pragma unroll
    for (int ni = 0; ni < 8; ++ni) {
      int col = hoff + ni * 16 + lm;
      float val = acc[ni][v] * inv * fm;
      size_t o = (size_t)row * DIM + col;
      if (mode == 0) Out[o] = val;
      else           Out[o] += val;
    }
  }
}

// ---------------------------------------------------------------------------
// Host side
// ---------------------------------------------------------------------------
extern "C" void kernel_launch(void* const* d_in, const int* in_sizes, int n_in,
                              void* d_out, int out_size, void* d_ws, size_t ws_size,
                              hipStream_t stream) {
  (void)in_sizes; (void)n_in; (void)out_size; (void)ws_size;

  const float* x     = (const float*)d_in[0];
  const float* ctx   = (const float*)d_in[1];
  const float* aud   = (const float*)d_in[2];
  const float* face  = (const float*)d_in[3];
  const int*   lens  = (const int*)d_in[4];
  const float* freqs = (const float*)d_in[8];
  const float* wq  = (const float*)d_in[9];  const float* bq  = (const float*)d_in[10];
  const float* wk  = (const float*)d_in[11]; const float* bk  = (const float*)d_in[12];
  const float* wv  = (const float*)d_in[13]; const float* bv  = (const float*)d_in[14];
  const float* wo  = (const float*)d_in[15]; const float* bo  = (const float*)d_in[16];
  const float* wki = (const float*)d_in[17]; const float* bki = (const float*)d_in[18];
  const float* wvi = (const float*)d_in[19]; const float* bvi = (const float*)d_in[20];
  const float* wka = (const float*)d_in[21]; const float* bka = (const float*)d_in[22];
  const float* wva = (const float*)d_in[23]; const float* bva = (const float*)d_in[24];
  const float* gq  = (const float*)d_in[25]; const float* gk  = (const float*)d_in[26];
  const float* gki = (const float*)d_in[27]; const float* gka = (const float*)d_in[28];
  float* out = (float*)d_out;

  size_t off = 0;
  auto alloc = [&](size_t bytes) -> char* {
    char* p = (char*)d_ws + off;
    off += (bytes + 255) & ~(size_t)255;
    return p;
  };

  const int LIMG_P = 288;   // img keys padded to multiple of 32
  const int LAUD   = 256;   // 8 frames * 32 audio keys

  __bf16* xb    = (__bf16*)alloc((size_t)L1 * DIM * 2);
  __bf16* ctxb  = (__bf16*)alloc((size_t)LCTX * DIM * 2);
  __bf16* audb  = (__bf16*)alloc((size_t)LAUD * 768 * 2);
  __bf16* wqT   = (__bf16*)alloc((size_t)DIM * DIM * 2);
  __bf16* wkT   = (__bf16*)alloc((size_t)DIM * DIM * 2);
  __bf16* wvT   = (__bf16*)alloc((size_t)DIM * DIM * 2);
  __bf16* woT   = (__bf16*)alloc((size_t)DIM * DIM * 2);
  __bf16* wkiT  = (__bf16*)alloc((size_t)DIM * DIM * 2);
  __bf16* wviT  = (__bf16*)alloc((size_t)DIM * DIM * 2);
  __bf16* wkaT  = (__bf16*)alloc((size_t)DIM * 768 * 2);
  __bf16* wvaT  = (__bf16*)alloc((size_t)DIM * 768 * 2);
  float*  q_raw    = (float*)alloc((size_t)L1 * DIM * 4);
  float*  ktxt_raw = (float*)alloc((size_t)LTXT * DIM * 4);
  float*  vtxt_raw = (float*)alloc((size_t)LTXT * DIM * 4);
  float*  kimg_raw = (float*)alloc((size_t)LIMG_P * DIM * 4);
  float*  vimg_raw = (float*)alloc((size_t)LIMG_P * DIM * 4);
  float*  kaud_raw = (float*)alloc((size_t)LAUD * DIM * 4);
  float*  vaud_raw = (float*)alloc((size_t)LAUD * DIM * 4);
  __bf16* qnb    = (__bf16*)alloc((size_t)L1 * DIM * 2);
  __bf16* qropeb = (__bf16*)alloc((size_t)L1 * DIM * 2);
  __bf16* ktxtb  = (__bf16*)alloc((size_t)LTXT * DIM * 2);
  __bf16* vtxtb  = (__bf16*)alloc((size_t)LTXT * DIM * 2);
  __bf16* kimgb  = (__bf16*)alloc((size_t)LIMG_P * DIM * 2);
  __bf16* vimgb  = (__bf16*)alloc((size_t)LIMG_P * DIM * 2);
  __bf16* kaudb  = (__bf16*)alloc((size_t)LAUD * DIM * 2);
  __bf16* vaudb  = (__bf16*)alloc((size_t)LAUD * DIM * 2);
  __bf16* vtxtT  = (__bf16*)alloc((size_t)DIM * LTXT * 2);
  __bf16* vimgT  = (__bf16*)alloc((size_t)DIM * LIMG_P * 2);
  __bf16* vaudT  = (__bf16*)alloc((size_t)DIM * LAUD * 2);
  float*  attn   = (float*)alloc((size_t)L1 * DIM * 4);
  __bf16* attnb  = (__bf16*)alloc((size_t)L1 * DIM * 2);

  // --- 1. convert activations to bf16 ---
  {
    long long n;
    n = (long long)L1 * DIM;
    cvt_f32_bf16_kernel<<<(unsigned)((n + 255) / 256), 256, 0, stream>>>(x, xb, n);
    n = (long long)LCTX * DIM;
    cvt_f32_bf16_kernel<<<(unsigned)((n + 255) / 256), 256, 0, stream>>>(ctx, ctxb, n);
    n = (long long)LAUD * 768;
    cvt_f32_bf16_kernel<<<(unsigned)((n + 255) / 256), 256, 0, stream>>>(aud, audb, n);
  }

  // --- 2. transpose weights (W[K][N] f32 -> WT[N][K] bf16) ---
  dim3 tb(32, 8);
  transpose_f32_bf16_kernel<<<dim3(64, 64), tb, 0, stream>>>(wq,  wqT,  DIM, DIM);
  transpose_f32_bf16_kernel<<<dim3(64, 64), tb, 0, stream>>>(wk,  wkT,  DIM, DIM);
  transpose_f32_bf16_kernel<<<dim3(64, 64), tb, 0, stream>>>(wv,  wvT,  DIM, DIM);
  transpose_f32_bf16_kernel<<<dim3(64, 64), tb, 0, stream>>>(wo,  woT,  DIM, DIM);
  transpose_f32_bf16_kernel<<<dim3(64, 64), tb, 0, stream>>>(wki, wkiT, DIM, DIM);
  transpose_f32_bf16_kernel<<<dim3(64, 64), tb, 0, stream>>>(wvi, wviT, DIM, DIM);
  transpose_f32_bf16_kernel<<<dim3(64, 24), tb, 0, stream>>>(wka, wkaT, 768, DIM);
  transpose_f32_bf16_kernel<<<dim3(64, 24), tb, 0, stream>>>(wva, wvaT, 768, DIM);

  // --- 3. projection GEMMs (f32 out) ---
  const __bf16* ctx_txt = ctxb + (size_t)LIMG * DIM;
  gemm_bias_kernel<<<dim3(64, 16), 256, 0, stream>>>(xb,      wqT,  bq,  q_raw,    L1,   DIM, DIM);
  gemm_bias_kernel<<<dim3(4, 16),  256, 0, stream>>>(ctx_txt, wkT,  bk,  ktxt_raw, LTXT, DIM, DIM);
  gemm_bias_kernel<<<dim3(4, 16),  256, 0, stream>>>(ctx_txt, wvT,  bv,  vtxt_raw, LTXT, DIM, DIM);
  gemm_bias_kernel<<<dim3(3, 16),  256, 0, stream>>>(ctxb,    wkiT, bki, kimg_raw, LIMG, DIM, DIM);
  gemm_bias_kernel<<<dim3(3, 16),  256, 0, stream>>>(ctxb,    wviT, bvi, vimg_raw, LIMG, DIM, DIM);
  gemm_bias_kernel<<<dim3(2, 16),  256, 0, stream>>>(audb,    wkaT, bka, kaud_raw, LAUD, DIM, 768);
  gemm_bias_kernel<<<dim3(2, 16),  256, 0, stream>>>(audb,    wvaT, bva, vaud_raw, LAUD, DIM, 768);

  // --- 4. RMSNorm / convert to bf16 (zero pad rows) ---
  rms_cvt_kernel<<<L1,     256, 0, stream>>>(q_raw,    qnb,   gq,  L1,   1);
  rms_cvt_kernel<<<LTXT,   256, 0, stream>>>(ktxt_raw, ktxtb, gk,  LTXT, 1);
  rms_cvt_kernel<<<LTXT,   256, 0, stream>>>(vtxt_raw, vtxtb, gk,  LTXT, 0);
  rms_cvt_kernel<<<LIMG_P, 256, 0, stream>>>(kimg_raw, kimgb, gki, LIMG, 1);
  rms_cvt_kernel<<<LIMG_P, 256, 0, stream>>>(vimg_raw, vimgb, gki, LIMG, 0);
  rms_cvt_kernel<<<LAUD,   256, 0, stream>>>(kaud_raw, kaudb, gka, LAUD, 1);
  rms_cvt_kernel<<<LAUD,   256, 0, stream>>>(vaud_raw, vaudb, gka, LAUD, 0);

  // --- 5. RoPE on q (for audio attention) ---
  rope_kernel<<<(L1 * NH * 64) / 256, 256, 0, stream>>>(qnb, freqs, qropeb);

  // --- 6. transpose V's to [DIM][keys] for contiguous PV B-fragments ---
  transpose_bf16_kernel<<<dim3(64, 16), tb, 0, stream>>>(vtxtb, vtxtT, LTXT,   DIM);
  transpose_bf16_kernel<<<dim3(64, 9),  tb, 0, stream>>>(vimgb, vimgT, LIMG_P, DIM);
  transpose_bf16_kernel<<<dim3(64, 8),  tb, 0, stream>>>(vaudb, vaudT, LAUD,   DIM);

  // --- 7. attentions (sequential accumulation into attn) ---
  dim3 ag(L1 / 16 / 4, NH);
  // txt (store), length-masked by context_lens
  flash_attn_kernel<<<ag, 128, 0, stream>>>(qnb, ktxtb, vtxtT, attn, nullptr, lens,
                                            LTXT, LTXT, LTXT, 0, 0);
  // img (add), 257 valid keys padded to 288
  flash_attn_kernel<<<ag, 128, 0, stream>>>(qnb, kimgb, vimgT, attn, nullptr, nullptr,
                                            LIMG, LIMG_P, LIMG_P, 1, 0);
  // audio (add * face_mask), per-frame 32 keys, roped q
  flash_attn_kernel<<<ag, 128, 0, stream>>>(qropeb, kaudb, vaudT, attn, face, nullptr,
                                            32, 32, LAUD, 2, 1);

  // --- 8. output projection ---
  {
    long long n = (long long)L1 * DIM;
    cvt_f32_bf16_kernel<<<(unsigned)((n + 255) / 256), 256, 0, stream>>>(attn, attnb, n);
  }
  gemm_bias_kernel<<<dim3(64, 16), 256, 0, stream>>>(attnb, woT, bo, out, L1, DIM, DIM);
}
